// QuantizedLlamaAttention_78151224918033
// MI455X (gfx1250) — 16-bit and 8-bit matrix operand order repaired and checked by reading, compiled, not run on hardware
//
#include <hip/hip_runtime.h>
#include <hip/hip_bf16.h>
#include <stdint.h>

typedef __attribute__((ext_vector_type(8)))  int       v8i;
typedef __attribute__((ext_vector_type(8)))  float     v8f;
typedef __attribute__((ext_vector_type(16))) _Float16  v16h;
typedef __attribute__((ext_vector_type(8)))  _Float16  v8h;

#define S_LEN   2048
#define H_DIM   4096
#define NHEADS  32
#define NKVH    8
#define HDIM    128
#define GQA     (NHEADS / NKVH)
#define QMAXF   127.0f
#define KSTG    128

__device__ __forceinline__ uint32_t lds_off32(const void* p) {
  return (uint32_t)(uintptr_t)p;
}
__device__ __forceinline__ void async_ld_b128(uint32_t lds, const void* gaddr) {
  asm volatile("global_load_async_to_lds_b128 %0, %1, off"
               :: "v"(lds), "v"(gaddr) : "memory");
}
__device__ __forceinline__ void wait_async0() {
  asm volatile("s_wait_asynccnt 0x0" ::: "memory");
}
__device__ __forceinline__ void wait_async3() {
  asm volatile("s_wait_asynccnt 0x3" ::: "memory");
}
__device__ __forceinline__ void wait_async8() {
  asm volatile("s_wait_asynccnt 0x8" ::: "memory");
}

__global__ __launch_bounds__(256) void quant_rows_kernel(
    const float* __restrict__ src, int8_t* __restrict__ dst,
    float* __restrict__ scale, int rowlen) {
  const int row = blockIdx.x;
  const int tid = threadIdx.x;
  const float* r = src + (size_t)row * rowlen;
  const float4* rv = (const float4*)r;
  const int nvec = rowlen >> 2;

  __shared__ float red[256];
  float am = 0.f;
  for (int i = tid; i < nvec; i += 256) {
    float4 x = rv[i];
    am = fmaxf(am, fmaxf(fmaxf(fabsf(x.x), fabsf(x.y)),
                         fmaxf(fabsf(x.z), fabsf(x.w))));
  }
  red[tid] = am;
  __syncthreads();
  for (int s = 128; s > 0; s >>= 1) {
    if (tid < s) red[tid] = fmaxf(red[tid], red[tid + s]);
    __syncthreads();
  }
  const float sc = fmaxf(red[0] / QMAXF, 1e-8f);
  if (tid == 0) scale[row] = sc;
  const float inv = 1.0f / sc;

  char4* dv = (char4*)(dst + (size_t)row * rowlen);
  for (int i = tid; i < nvec; i += 256) {
    float4 x = rv[i];
    char4 q;
    q.x = (char)fminf(fmaxf(rintf(x.x * inv), -128.f), 127.f);
    q.y = (char)fminf(fmaxf(rintf(x.y * inv), -128.f), 127.f);
    q.z = (char)fminf(fmaxf(rintf(x.z * inv), -128.f), 127.f);
    q.w = (char)fminf(fmaxf(rintf(x.w * inv), -128.f), 127.f);
    dv[i] = q;
  }
}

__global__ __launch_bounds__(256) void gemm_i8_kernel(
    const int8_t* __restrict__ A, const int8_t* __restrict__ B,
    const float* __restrict__ sa, const float* __restrict__ sb,
    float* __restrict__ C, int M, int N, int K) {
  __shared__ int8_t lA[2][32 * KSTG];
  __shared__ int8_t lB[2][64 * KSTG];

  const int tid   = threadIdx.x;
  const int lane  = tid & 31;
  const int wave  = tid >> 5;
  const int lhalf = lane >> 4;
  const int lcol  = lane & 15;
  const int mW = (wave >> 2) * 16;
  const int nW = (wave & 3) * 16;
  const int mB = blockIdx.x * 32;
  const int nB = blockIdx.y * 64;

  const int ar = tid >> 3, ac = (tid & 7) * 16;
  const int br = tid >> 2, bc = (tid & 3) * 32;
  const int8_t* gA = A + (size_t)(mB + ar) * K + ac;
  const int8_t* gB = B + (size_t)(nB + br) * K + bc;

  auto issue_stage = [&](int st, int k) {
    async_ld_b128(lds_off32(&lA[st][ar * KSTG + ac]), gA + k);
    async_ld_b128(lds_off32(&lB[st][br * KSTG + bc]), gB + k);
    async_ld_b128(lds_off32(&lB[st][br * KSTG + bc + 16]), gB + k + 16);
  };

  v8i acc = {};
  auto compute_stage = [&](int st) {
    const int8_t* pA = &lA[st][(mW + lcol) * KSTG];
    const int8_t* pB = &lB[st][(nW + lcol) * KSTG];
#pragma unroll
    for (int ks = 0; ks < KSTG; ks += 64) {
      union { v8i v; int4 q[2]; } ua;
      const int8_t* ap = pA + ks + lhalf * 16;
      ua.q[0] = *(const int4*)(ap);
      ua.q[1] = *(const int4*)(ap + 32);

      union { v8i v; int4 q[2]; } ub;
      const int8_t* bp = pB + ks + lhalf * 16;
      ub.q[0] = *(const int4*)(bp);
      ub.q[1] = *(const int4*)(bp + 32);

      acc = __builtin_amdgcn_wmma_i32_16x16x64_iu8(
           true, ua.v,  true, ub.v, acc,
           false,  false);
    }
  };

  const int nStages = K / KSTG;
  issue_stage(0, 0);
  for (int s = 0; s < nStages - 1; ++s) {
    issue_stage((s + 1) & 1, (s + 1) * KSTG);
    wait_async3();
    __syncthreads();
    compute_stage(s & 1);
    __syncthreads();
  }
  wait_async0();
  __syncthreads();
  compute_stage((nStages - 1) & 1);

  const int mT = mB + mW, nT = nB + nW;
#pragma unroll
  for (int v = 0; v < 8; ++v) {
    const int row = mT + lhalf * 8 + v;
    const int col = nT + lcol;
    C[(size_t)row * N + col] = (float)acc[v] * sa[row] * sb[col];
  }
}

__global__ __launch_bounds__(64) void rope_pack_f16_kernel(
    const float* __restrict__ src, const float* __restrict__ cosb,
    const float* __restrict__ sinb, _Float16* __restrict__ dst, int nheads) {
  const int s = blockIdx.x / nheads;
  const int h = blockIdx.x % nheads;
  const int d = threadIdx.x;
  const size_t ib = ((size_t)s * nheads + h) * HDIM;
  const float x0 = src[ib + d];
  const float x1 = src[ib + d + 64];
  const float c0 = cosb[(size_t)s * HDIM + d];
  const float s0 = sinb[(size_t)s * HDIM + d];
  const float c1 = cosb[(size_t)s * HDIM + d + 64];
  const float s1 = sinb[(size_t)s * HDIM + d + 64];
  const float y0 = x0 * c0 - x1 * s0;
  const float y1 = x1 * c1 + x0 * s1;
  const size_t ob = ((size_t)h * S_LEN + s) * HDIM;
  dst[ob + d]      = (_Float16)y0;
  dst[ob + d + 64] = (_Float16)y1;
}

__global__ __launch_bounds__(128) void v_transpose_f16_kernel(
    const float* __restrict__ src, _Float16* __restrict__ dst) {
  const int s = blockIdx.x / NKVH;
  const int h = blockIdx.x % NKVH;
  const int d = threadIdx.x;
  const float x = src[((size_t)s * NKVH + h) * HDIM + d];
  dst[((size_t)h * HDIM + d) * S_LEN + s] = (_Float16)x;
}

__global__ __launch_bounds__(128) void flash_attn_kernel(
    const _Float16* __restrict__ qh,
    const _Float16* __restrict__ kh,
    const _Float16* __restrict__ vt,
    float* __restrict__ attn) {
  const int h      = blockIdx.y;
  const int kvh    = h / GQA;
  const int qBlock = blockIdx.x * 64;
  const int tid    = threadIdx.x;
  const int wave   = tid >> 5;
  const int lane   = tid & 31;
  const int lhalf  = lane >> 4;
  const int lcol   = lane & 15;
  const int qw     = qBlock + wave * 16;
  const float scl  = 0.08838834764831845f;

  __shared__ _Float16 ldsK[2][32 * HDIM];
  __shared__ _Float16 ldsV[2][HDIM * 32];
  __shared__ _Float16 plds[4][16 * 32];

  const int kr = tid >> 2;
  const int kc = (tid & 3) * 64;
  const int vr = tid;

  auto issue_chunk = [&](int kcnk, int st) {
    const int k0 = kcnk * 32;
    const char* gK = (const char*)(kh + ((size_t)kvh * S_LEN + k0 + kr) * HDIM) + kc;
    char*       lK = (char*)&ldsK[st][kr * HDIM] + kc;
    const char* gV = (const char*)(vt + ((size_t)kvh * HDIM + vr) * S_LEN + k0);
    char*       lV = (char*)&ldsV[st][vr * 32];
#pragma unroll
    for (int b = 0; b < 64; b += 16) {
      async_ld_b128(lds_off32(lK + b), gK + b);
      async_ld_b128(lds_off32(lV + b), gV + b);
    }
  };

  const v8f zf = {};
  float mrun[8], lrun[8];
  v8f acco[8];
#pragma unroll
  for (int i = 0; i < 8; ++i) { mrun[i] = -3.0e38f; lrun[i] = 0.f; acco[i] = zf; }

  const _Float16* qbase = qh + ((size_t)h * S_LEN + qw + lcol) * HDIM;

  auto process_chunk = [&](int kcnk, int st) {
    const int k0 = kcnk * 32;

    v8f sacc0 = zf, sacc1 = zf;
#pragma unroll
    for (int dstep = 0; dstep < 4; ++dstep) {
      union { v16h v; v8h p[2]; } aq;
      const _Float16* qp = qbase + dstep * 32 + lhalf * 16;
      aq.p[0] = *(const v8h*)(qp);
      aq.p[1] = *(const v8h*)(qp + 8);

      v16h bk0 = *(const v16h*)&ldsK[st][(lcol)      * HDIM + dstep * 32 + lhalf * 16];
      v16h bk1 = *(const v16h*)&ldsK[st][(16 + lcol) * HDIM + dstep * 32 + lhalf * 16];
      sacc0 = __builtin_amdgcn_wmma_f32_16x16x32_f16(false, aq.v, false, bk0,
                                                     (short)0, sacc0, false, false);
      sacc1 = __builtin_amdgcn_wmma_f32_16x16x32_f16(false, aq.v, false, bk1,
                                                     (short)0, sacc1, false, false);
    }

    float pv0[8], pv1[8], cmax[8];
#pragma unroll
    for (int v = 0; v < 8; ++v) {
      const int qpos = qw + lhalf * 8 + v;
      float v0 = sacc0[v] * scl;
      float v1 = sacc1[v] * scl;
      if (k0 + lcol > qpos)      v0 = -3.0e38f;
      if (k0 + 16 + lcol > qpos) v1 = -3.0e38f;
      pv0[v] = v0; pv1[v] = v1;
      cmax[v] = fmaxf(v0, v1);
    }
#pragma unroll
    for (int m = 1; m < 16; m <<= 1) {
#pragma unroll
      for (int v = 0; v < 8; ++v)
        cmax[v] = fmaxf(cmax[v], __shfl_xor(cmax[v], m, 32));
    }
    float alpha[8], psum[8];
#pragma unroll
    for (int v = 0; v < 8; ++v) {
      const float mnew = fmaxf(mrun[v], cmax[v]);
      alpha[v] = __expf(mrun[v] - mnew);
      mrun[v]  = mnew;
      float p0 = (pv0[v] <= -1.0e38f) ? 0.f : __expf(pv0[v] - mnew);
      float p1 = (pv1[v] <= -1.0e38f) ? 0.f : __expf(pv1[v] - mnew);
      pv0[v] = p0; pv1[v] = p1;
      psum[v] = p0 + p1;
    }
#pragma unroll
    for (int m = 1; m < 16; m <<= 1) {
#pragma unroll
      for (int v = 0; v < 8; ++v)
        psum[v] += __shfl_xor(psum[v], m, 32);
    }
#pragma unroll
    for (int v = 0; v < 8; ++v) lrun[v] = lrun[v] * alpha[v] + psum[v];
#pragma unroll
    for (int dt = 0; dt < 8; ++dt) {
#pragma unroll
      for (int v = 0; v < 8; ++v) acco[dt][v] *= alpha[v];
    }

#pragma unroll
    for (int v = 0; v < 8; ++v) {
      const int prow = v + lhalf * 8;
      plds[wave][prow * 32 + lcol]      = (_Float16)pv0[v];
      plds[wave][prow * 32 + 16 + lcol] = (_Float16)pv1[v];
    }
    __syncthreads();

    union { v16h v; v8h p[2]; } ap;
    ap.p[0] = *(const v8h*)&plds[wave][lcol * 32 + lhalf * 16];
    ap.p[1] = *(const v8h*)&plds[wave][lcol * 32 + lhalf * 16 + 8];

#pragma unroll
    for (int dt = 0; dt < 8; ++dt) {
      v16h bv = *(const v16h*)&ldsV[st][(dt * 16 + lcol) * 32 + lhalf * 16];
      acco[dt] = __builtin_amdgcn_wmma_f32_16x16x32_f16(false, ap.v, false, bv,
                                                        (short)0, acco[dt],
                                                        false, false);
    }
    __syncthreads();
  };

  const int nChunks = (qBlock + 64) / 32;
  issue_chunk(0, 0);
  for (int kcnk = 0; kcnk < nChunks - 1; ++kcnk) {
    issue_chunk(kcnk + 1, (kcnk + 1) & 1);
    wait_async8();
    __syncthreads();
    process_chunk(kcnk, kcnk & 1);
  }
  wait_async0();
  __syncthreads();
  process_chunk(nChunks - 1, (nChunks - 1) & 1);

#pragma unroll
  for (int dt = 0; dt < 8; ++dt) {
#pragma unroll
    for (int v = 0; v < 8; ++v) {
      const int row = qw + lhalf * 8 + v;
      attn[(size_t)row * (NHEADS * HDIM) + h * HDIM + dt * 16 + lcol] =
          acco[dt][v] / lrun[v];
    }
  }
}

extern "C" void kernel_launch(void* const* d_in, const int* in_sizes, int n_in,
                              void* d_out, int out_size, void* d_ws, size_t ws_size,
                              hipStream_t stream) {
  (void)in_sizes; (void)n_in; (void)out_size; (void)ws_size;
  const float* x    = (const float*)d_in[0];
  const float* cosb = (const float*)d_in[1];
  const float* sinb = (const float*)d_in[2];
  const float* Wq   = (const float*)d_in[3];
  const float* Wk   = (const float*)d_in[4];
  const float* Wv   = (const float*)d_in[5];
  const float* Wo   = (const float*)d_in[6];
  float* out = (float*)d_out;

  char* ws = (char*)d_ws;
  size_t off = 0;
  auto alloc = [&](size_t bytes) -> char* {
    char* p = ws + off;
    off = (off + bytes + 255) & ~(size_t)255;
    return p;
  };

  int8_t* xq   = (int8_t*)alloc((size_t)S_LEN * H_DIM);
  int8_t* wqq  = (int8_t*)alloc((size_t)H_DIM * H_DIM);
  int8_t* wkq  = (int8_t*)alloc((size_t)(NKVH * HDIM) * H_DIM);
  int8_t* wvq  = (int8_t*)alloc((size_t)(NKVH * HDIM) * H_DIM);
  int8_t* woq  = (int8_t*)alloc((size_t)H_DIM * H_DIM);
  float* sx    = (float*)alloc(S_LEN * 4);
  float* swq   = (float*)alloc(H_DIM * 4);
  float* swk   = (float*)alloc(NKVH * HDIM * 4);
  float* swv   = (float*)alloc(NKVH * HDIM * 4);
  float* swo   = (float*)alloc(H_DIM * 4);
  float* sattn = (float*)alloc(S_LEN * 4);
  float* qf32  = (float*)alloc((size_t)S_LEN * H_DIM * 4);
  float* kf32  = (float*)alloc((size_t)S_LEN * NKVH * HDIM * 4);
  float* vf32  = (float*)alloc((size_t)S_LEN * NKVH * HDIM * 4);
  _Float16* qh = (_Float16*)alloc((size_t)S_LEN * H_DIM * 2);
  _Float16* kh = (_Float16*)alloc((size_t)S_LEN * NKVH * HDIM * 2);
  _Float16* vt = (_Float16*)alloc((size_t)S_LEN * NKVH * HDIM * 2);
  float*  attn  = qf32;
  int8_t* attnq = (int8_t*)kf32;

  quant_rows_kernel<<<S_LEN, 256, 0, stream>>>(x,  xq,  sx,  H_DIM);
  quant_rows_kernel<<<H_DIM, 256, 0, stream>>>(Wq, wqq, swq, H_DIM);
  quant_rows_kernel<<<NKVH * HDIM, 256, 0, stream>>>(Wk, wkq, swk, H_DIM);
  quant_rows_kernel<<<NKVH * HDIM, 256, 0, stream>>>(Wv, wvq, swv, H_DIM);
  quant_rows_kernel<<<H_DIM, 256, 0, stream>>>(Wo, woq, swo, H_DIM);

  gemm_i8_kernel<<<dim3(S_LEN / 32, H_DIM / 64), 256, 0, stream>>>(
      xq, wqq, sx, swq, qf32, S_LEN, H_DIM, H_DIM);
  gemm_i8_kernel<<<dim3(S_LEN / 32, (NKVH * HDIM) / 64), 256, 0, stream>>>(
      xq, wkq, sx, swk, kf32, S_LEN, NKVH * HDIM, H_DIM);
  gemm_i8_kernel<<<dim3(S_LEN / 32, (NKVH * HDIM) / 64), 256, 0, stream>>>(
      xq, wvq, sx, swv, vf32, S_LEN, NKVH * HDIM, H_DIM);

  rope_pack_f16_kernel<<<S_LEN * NHEADS, 64, 0, stream>>>(qf32, cosb, sinb, qh, NHEADS);
  rope_pack_f16_kernel<<<S_LEN * NKVH, 64, 0, stream>>>(kf32, cosb, sinb, kh, NKVH);
  v_transpose_f16_kernel<<<S_LEN * NKVH, 128, 0, stream>>>(vf32, vt);

  flash_attn_kernel<<<dim3(S_LEN / 64, NHEADS), 128, 0, stream>>>(qh, kh, vt, attn);

  quant_rows_kernel<<<S_LEN, 256, 0, stream>>>(attn, attnq, sattn, H_DIM);
  gemm_i8_kernel<<<dim3(S_LEN / 32, H_DIM / 64), 256, 0, stream>>>(
      attnq, woq, sattn, swo, out, S_LEN, H_DIM, H_DIM);
}
